// Conv2Seq_39754217292118
// MI455X (gfx1250) — matrix operand order repaired and checked by reading, not compiled, not run on hardware
//
#include <hip/hip_runtime.h>

// ---------------------------------------------------------------------------
// Model dims
// ---------------------------------------------------------------------------
#define BATCH 32
#define TSEQ  2048
#define DIN   64
#define HDIM  1024
#define DOUT  64
#define NSTEP 60
#define T1    2045   // T - 4 + 1
#define T2    2038   // T1 - 8 + 1
#define LSEQ  2023   // T2 - 16 + 1
#define KC1   1120   // 32 + 64 + 1024
#define KC23  2080   // 32 + 1024 + 1024
#define NGATE 4096
#define NSPLIT 4     // K-split factor for the gate GEMMs

typedef __attribute__((ext_vector_type(16))) __bf16 v16bf;
typedef __attribute__((ext_vector_type(8)))  __bf16 bf16x8;
typedef __attribute__((ext_vector_type(8)))  float  v8f;

__device__ __forceinline__ __bf16 f2bf(float f) {
    unsigned u = __builtin_bit_cast(unsigned, f);
    unsigned r = u + 0x7FFFu + ((u >> 16) & 1u);          // round-to-nearest-even
    if ((u & 0x7F800000u) == 0x7F800000u) r = u;          // keep inf/nan
    return __builtin_bit_cast(__bf16, (unsigned short)(r >> 16));
}
__device__ __forceinline__ float sigf(float x) { return 1.0f / (1.0f + __expf(-x)); }

// ---------------------------------------------------------------------------
// Conv stack (one-time, tiny FLOPs)
// ---------------------------------------------------------------------------
__global__ void __launch_bounds__(256) conv1_kernel(
    const float* __restrict__ x, const float* __restrict__ w,
    const float* __restrict__ bias, float* __restrict__ y) {
    int idx = blockIdx.x * 256 + threadIdx.x;
    if (idx >= BATCH * 8 * T1) return;
    int t = idx % T1, o = (idx / T1) % 8, b = idx / (T1 * 8);
    const float* xb = x + (long)b * TSEQ * DIN;
    const float* wo = w + o * DIN * 4;
    float s = bias[o];
    for (int i = 0; i < DIN; ++i)
        for (int k = 0; k < 4; ++k)
            s += xb[(t + k) * DIN + i] * wo[i * 4 + k];
    y[idx] = fmaxf(s, 0.0f);
}

__global__ void __launch_bounds__(256) conv2_kernel(
    const float* __restrict__ e1, const float* __restrict__ w,
    const float* __restrict__ bias, float* __restrict__ y) {
    int idx = blockIdx.x * 256 + threadIdx.x;
    if (idx >= BATCH * 16 * T2) return;
    int t = idx % T2, o = (idx / T2) % 16, b = idx / (T2 * 16);
    const float* eb = e1 + (long)b * 8 * T1;
    const float* wo = w + o * 8 * 8;
    float s = bias[o];
    for (int i = 0; i < 8; ++i)
        for (int k = 0; k < 8; ++k)
            s += eb[i * T1 + t + k] * wo[i * 8 + k];
    y[idx] = fmaxf(s, 0.0f);
}

// enc stored (B, L, 32) row-major for attention locality
__global__ void __launch_bounds__(256) conv3_kernel(
    const float* __restrict__ e2, const float* __restrict__ w,
    const float* __restrict__ bias, float* __restrict__ enc) {
    int idx = blockIdx.x * 256 + threadIdx.x;
    if (idx >= BATCH * LSEQ * 32) return;
    int o = idx % 32, t = (idx / 32) % LSEQ, b = idx / (32 * LSEQ);
    const float* eb = e2 + (long)b * 16 * T2;
    const float* wo = w + o * 16 * 16;
    float s = bias[o];
    for (int i = 0; i < 16; ++i)
        for (int k = 0; k < 16; ++k)
            s += eb[i * T2 + t + k] * wo[i * 16 + k];
    enc[idx] = fmaxf(s, 0.0f);
}

// ---------------------------------------------------------------------------
// Pack [Wih ; Whh] -> bf16 concatenated (NGATE x kcat) row-major
// ---------------------------------------------------------------------------
__global__ void __launch_bounds__(256) pack_wcat_kernel(
    const float* __restrict__ wih, int kih, const float* __restrict__ whh,
    __bf16* __restrict__ dst, int kcat) {
    long n = (long)NGATE * kcat;
    for (long i = blockIdx.x * 256L + threadIdx.x; i < n; i += 256L * gridDim.x) {
        long row = i / kcat, col = i % kcat;
        float v = (col < kih) ? wih[row * kih + col] : whh[row * HDIM + (col - kih)];
        dst[i] = f2bf(v);
    }
}

// ---------------------------------------------------------------------------
// Init: zero h/c states; zero act buffers; seed act1 "prev" slot with x[:,-1,:]
// ---------------------------------------------------------------------------
__global__ void __launch_bounds__(256) init_state_kernel(
    const float* __restrict__ x, float* __restrict__ states,
    __bf16* __restrict__ act1, __bf16* __restrict__ act2, __bf16* __restrict__ act3) {
    long tid = blockIdx.x * 256L + threadIdx.x;
    long stride = 256L * gridDim.x;
    for (long i = tid; i < 6L * BATCH * HDIM; i += stride) states[i] = 0.0f;
    for (long i = tid; i < (long)BATCH * KC1; i += stride) {
        long b = i / KC1, c = i % KC1;
        float v = 0.0f;
        if (c >= 32 && c < 96) v = x[(b * TSEQ + (TSEQ - 1)) * DIN + (c - 32)];
        act1[i] = f2bf(v);
    }
    for (long i = tid; i < (long)BATCH * KC23; i += stride) act2[i] = f2bf(0.0f);
    for (long i = tid; i < (long)BATCH * KC23; i += stride) act3[i] = f2bf(0.0f);
}

// ---------------------------------------------------------------------------
// Attention, keys never materialized:
//   q = Wa^T h  (32-dim);  score[l] = q . enc[b,l,:];  softmax;  ctx = sum p*enc
// (h . b_a is constant over l -> cancels in softmax)
// Block (j, b), 256 threads.
// ---------------------------------------------------------------------------
__global__ void __launch_bounds__(256) attend_kernel(
    const float* __restrict__ enc,
    const float* __restrict__ h1, const float* __restrict__ h2, const float* __restrict__ h3,
    const float* __restrict__ wa1, const float* __restrict__ wa2, const float* __restrict__ wa3,
    __bf16* __restrict__ act1, __bf16* __restrict__ act2, __bf16* __restrict__ act3) {
    __shared__ float hs[HDIM];
    __shared__ float sc[LSEQ];
    __shared__ float red[256];
    __shared__ float qv[32];

    const int j = blockIdx.x, b = blockIdx.y, t = threadIdx.x;
    const float* h  = (j == 0 ? h1 : (j == 1 ? h2 : h3)) + (long)b * HDIM;
    const float* wa = (j == 0 ? wa1 : (j == 1 ? wa2 : wa3));
    __bf16* act; int kc;
    if (j == 0)      { act = act1; kc = KC1;  }
    else if (j == 1) { act = act2; kc = KC23; }
    else             { act = act3; kc = KC23; }

    for (int i = t; i < HDIM; i += 256) hs[i] = h[i];
    __syncthreads();

    // q[d] = sum_hh hs[hh] * wa[hh*32 + d]
    {
        const int d = t & 31, seg = t >> 5;
        float p = 0.0f;
        for (int i = 0; i < 128; ++i) {
            int hh = seg * 128 + i;
            p += hs[hh] * wa[hh * 32 + d];
        }
        red[t] = p;
        __syncthreads();
        if (t < 32) {
            float s = 0.0f;
            for (int k = 0; k < 8; ++k) s += red[t + k * 32];
            qv[t] = s;
        }
        __syncthreads();
    }

    const float* eb = enc + (long)b * LSEQ * 32;
    float lmax = -3.0e38f;
    for (int l = t; l < LSEQ; l += 256) {
        const float* e = eb + l * 32;
        float s = 0.0f;
#pragma unroll
        for (int d = 0; d < 32; ++d) s += qv[d] * e[d];
        sc[l] = s;
        lmax = fmaxf(lmax, s);
    }
    red[t] = lmax;
    __syncthreads();
    for (int o = 128; o > 0; o >>= 1) { if (t < o) red[t] = fmaxf(red[t], red[t + o]); __syncthreads(); }
    float mx = red[0];
    __syncthreads();

    float lsum = 0.0f;
    for (int l = t; l < LSEQ; l += 256) { float p = __expf(sc[l] - mx); sc[l] = p; lsum += p; }
    red[t] = lsum;
    __syncthreads();
    for (int o = 128; o > 0; o >>= 1) { if (t < o) red[t] += red[t + o]; __syncthreads(); }
    float Z = red[0];
    __syncthreads();

    // ctx[d] = (1/Z) sum_l p[l] * enc[b,l,d]  -> bf16 into act[b*kc + d]
    {
        const int d = t & 31, seg = t >> 5;
        float p = 0.0f;
        for (int l = seg; l < LSEQ; l += 8) p += sc[l] * eb[l * 32 + d];
        red[t] = p;
        __syncthreads();
        if (t < 32) {
            float s = 0.0f;
            for (int k = 0; k < 8; ++k) s += red[t + k * 32];
            act[(long)b * kc + t] = f2bf(s / Z);
        }
    }
}

// ---------------------------------------------------------------------------
// Fragment load helper (CDNA5 ISA 16-bit A/B VGPR layouts).
// The element copies are register-coalesced by the backend: each v16bf is
// materialized directly in the destination pairs of two global_load_b128s.
// ---------------------------------------------------------------------------
__device__ __forceinline__ void load_frags(
    const __bf16* pa0, const __bf16* pa1,
    const __bf16* pw0, const __bf16* pw1,
    v16bf& aLo, v16bf& aHi, v16bf& bLo, v16bf& bHi) {
    bf16x8 a00 = *(const bf16x8*)(pa0);
    bf16x8 a01 = *(const bf16x8*)(pa0 + 16);
    bf16x8 a10 = *(const bf16x8*)(pa1);
    bf16x8 a11 = *(const bf16x8*)(pa1 + 16);
    bf16x8 b00 = *(const bf16x8*)(pw0);
    bf16x8 b01 = *(const bf16x8*)(pw0 + 16);
    bf16x8 b10 = *(const bf16x8*)(pw1);
    bf16x8 b11 = *(const bf16x8*)(pw1 + 16);
#pragma unroll
    for (int i = 0; i < 8; ++i) {
        aLo[i] = a00[i]; aLo[i + 8] = a01[i];
        aHi[i] = a10[i]; aHi[i + 8] = a11[i];
        bLo[i] = b00[i]; bLo[i + 8] = b01[i];
        bHi[i] = b10[i]; bHi[i + 8] = b11[i];
    }
}

// ---------------------------------------------------------------------------
// WMMA GEMM with K-split:
//   Gpart[split] (32 x 4096) += A[32, Kslice] * W[N, Kslice]^T   (bf16 in, f32 acc)
// Each wave owns a 32x32 output tile: 2 A-frags x 2 B-frags -> 4 WMMAs per
// K-step sharing fragments. Manually double-buffered (next K-step's 8 b128
// loads issued before the current 4 WMMAs) and unrolled x2 so the buffer
// rotation is eliminated by copy propagation (no v_mov churn).
// grid = (NGATE/32, NSPLIT), block = 32 (one wave).
// ---------------------------------------------------------------------------
__global__ void __launch_bounds__(32) gemm_wmma_kernel(
    const __bf16* __restrict__ A, const __bf16* __restrict__ W,
    float* __restrict__ G, int K) {
    const int lane  = threadIdx.x;
    const int n0    = blockIdx.x * 32;
    const int split = blockIdx.y;
    const int half  = lane >> 4;     // 0 or 1
    const int l15   = lane & 15;

    const int iters = K / 32;                          // 35 or 65
    const int per   = (iters + NSPLIT - 1) / NSPLIT;
    const int i0    = split * per;
    int i1          = i0 + per; if (i1 > iters) i1 = iters;

    // A (16x32 bf16): lane row = l15 (+16 for hi tile); elems e<8 -> k = e + 8*half,
    //                 elems e>=8 -> k = 16 + (e-8) + 8*half
    const __bf16* pa0 = A + (long)(l15)      * K + half * 8  + (long)i0 * 32;  // rows 0..15
    const __bf16* pa1 = A + (long)(16 + l15) * K + half * 8  + (long)i0 * 32;  // rows 16..31
    // B (32x16 bf16): lane col = n + l15; same k order as A (e<8 -> k = e + 8*half, e>=8 -> k = 16 + (e-8) + 8*half)
    const __bf16* pw0 = W + (long)(n0 + l15)      * K + half * 8 + (long)i0 * 32;
    const __bf16* pw1 = W + (long)(n0 + 16 + l15) * K + half * 8 + (long)i0 * 32;

    v8f acc00 = {}, acc01 = {}, acc10 = {}, acc11 = {};
    if (i0 < i1) {
        v16bf aLo, aHi, bLo, bHi;
        load_frags(pa0, pa1, pw0, pw1, aLo, aHi, bLo, bHi);
        const int nleft = i1 - i0 - 1;
#pragma unroll 2
        for (int it = 0; it < nleft; ++it) {
            pa0 += 32; pa1 += 32; pw0 += 32; pw1 += 32;
            __builtin_prefetch(pw0 + 128, 0, 1);   // global_prefetch_b8, ~4 iters ahead
            __builtin_prefetch(pw1 + 128, 0, 1);
            v16bf naLo, naHi, nbLo, nbHi;
            load_frags(pa0, pa1, pw0, pw1, naLo, naHi, nbLo, nbHi);
            acc00 = __builtin_amdgcn_wmma_f32_16x16x32_bf16(false, aLo, false, bLo, (short)0, acc00, false, false);
            acc01 = __builtin_amdgcn_wmma_f32_16x16x32_bf16(false, aLo, false, bHi, (short)0, acc01, false, false);
            acc10 = __builtin_amdgcn_wmma_f32_16x16x32_bf16(false, aHi, false, bLo, (short)0, acc10, false, false);
            acc11 = __builtin_amdgcn_wmma_f32_16x16x32_bf16(false, aHi, false, bHi, (short)0, acc11, false, false);
            aLo = naLo; aHi = naHi; bLo = nbLo; bHi = nbHi;
        }
        acc00 = __builtin_amdgcn_wmma_f32_16x16x32_bf16(false, aLo, false, bLo, (short)0, acc00, false, false);
        acc01 = __builtin_amdgcn_wmma_f32_16x16x32_bf16(false, aLo, false, bHi, (short)0, acc01, false, false);
        acc10 = __builtin_amdgcn_wmma_f32_16x16x32_bf16(false, aHi, false, bLo, (short)0, acc10, false, false);
        acc11 = __builtin_amdgcn_wmma_f32_16x16x32_bf16(false, aHi, false, bHi, (short)0, acc11, false, false);
    }

    // C/D f32 16x16: VGPR r -> row mBase + r + 8*half, col nBase + l15
    float* gp = G + (long)split * BATCH * NGATE;
    float* g00 = gp + (long)(half * 8)      * NGATE + n0 + l15;
    float* g01 = g00 + 16;
    float* g10 = gp + (long)(16 + half * 8) * NGATE + n0 + l15;
    float* g11 = g10 + 16;
#pragma unroll
    for (int r = 0; r < 8; ++r) {
        g00[(long)r * NGATE] = acc00[r];
        g01[(long)r * NGATE] = acc01[r];
        g10[(long)r * NGATE] = acc10[r];
        g11[(long)r * NGATE] = acc11[r];
    }
}

// ---------------------------------------------------------------------------
// LSTM pointwise: sum K-split gate partials + biases -> (c,h);
// scatter bf16(h) into next-layer / recurrent slots
// ---------------------------------------------------------------------------
__global__ void __launch_bounds__(256) lstm_update_kernel(
    const float* __restrict__ G, const float* __restrict__ bih, const float* __restrict__ bhh,
    float* __restrict__ hst, float* __restrict__ cst,
    __bf16* __restrict__ dA, long strideA, int offA,
    __bf16* __restrict__ dB, long strideB, int offB) {
    int idx = blockIdx.x * 256 + threadIdx.x;    // BATCH*HDIM = 32768
    int b = idx >> 10, u = idx & 1023;
    const float* g = G + (long)b * NGATE;
    float gi = bih[u]        + bhh[u];
    float gf = bih[1024 + u] + bhh[1024 + u];
    float gg = bih[2048 + u] + bhh[2048 + u];
    float go = bih[3072 + u] + bhh[3072 + u];
#pragma unroll
    for (int sp = 0; sp < NSPLIT; ++sp) {
        const float* gs = g + (long)sp * BATCH * NGATE;
        gi += gs[u];
        gf += gs[1024 + u];
        gg += gs[2048 + u];
        go += gs[3072 + u];
    }
    float c = sigf(gf) * cst[idx] + sigf(gi) * tanhf(gg);
    float h = sigf(go) * tanhf(c);
    cst[idx] = c;
    hst[idx] = h;
    __bf16 hb = f2bf(h);
    dA[(long)b * strideA + offA + u] = hb;
    if (dB) dB[(long)b * strideB + offB + u] = hb;
}

// ---------------------------------------------------------------------------
// y = h3 @ w_out^T + b_out ; write to output (B, STEPS, DOUT) and act1 "prev"
// ---------------------------------------------------------------------------
__global__ void __launch_bounds__(64) out_kernel(
    const float* __restrict__ h3, const float* __restrict__ wout,
    const float* __restrict__ bout, float* __restrict__ out,
    __bf16* __restrict__ act1, int step) {
    int b = blockIdx.x, o = threadIdx.x;
    const float* h = h3 + (long)b * HDIM;
    const float* w = wout + (long)o * HDIM;
    float s = bout[o];
    for (int k = 0; k < HDIM; ++k) s += h[k] * w[k];
    out[((long)b * NSTEP + step) * DOUT + o] = s;
    act1[(long)b * KC1 + 32 + o] = f2bf(s);
}

// ---------------------------------------------------------------------------
// Host driver
// ---------------------------------------------------------------------------
extern "C" void kernel_launch(void* const* d_in, const int* in_sizes, int n_in,
                              void* d_out, int out_size, void* d_ws, size_t ws_size,
                              hipStream_t stream) {
    const float* x    = (const float*)d_in[0];
    const float* w_c1 = (const float*)d_in[1];  const float* b_c1 = (const float*)d_in[2];
    const float* w_c2 = (const float*)d_in[3];  const float* b_c2 = (const float*)d_in[4];
    const float* w_c3 = (const float*)d_in[5];  const float* b_c3 = (const float*)d_in[6];
    const float* w_a1 = (const float*)d_in[7];
    const float* w_a2 = (const float*)d_in[9];
    const float* w_a3 = (const float*)d_in[11];
    const float* wih1 = (const float*)d_in[13]; const float* whh1 = (const float*)d_in[14];
    const float* bih1 = (const float*)d_in[15]; const float* bhh1 = (const float*)d_in[16];
    const float* wih2 = (const float*)d_in[17]; const float* whh2 = (const float*)d_in[18];
    const float* bih2 = (const float*)d_in[19]; const float* bhh2 = (const float*)d_in[20];
    const float* wih3 = (const float*)d_in[21]; const float* whh3 = (const float*)d_in[22];
    const float* bih3 = (const float*)d_in[23]; const float* bhh3 = (const float*)d_in[24];
    const float* wout = (const float*)d_in[25]; const float* bout = (const float*)d_in[26];
    float* out = (float*)d_out;

    char* base = (char*)d_ws;
    size_t off = 0;
    auto grab = [&](size_t bytes) -> void* {
        off = (off + 255) & ~(size_t)255;
        void* p = base + off;
        off += bytes;
        return p;
    };
    float*  e1     = (float*)grab((size_t)BATCH * 8  * T1   * 4);
    float*  e2     = (float*)grab((size_t)BATCH * 16 * T2   * 4);
    float*  enc    = (float*)grab((size_t)BATCH * LSEQ * 32 * 4);
    float*  states = (float*)grab((size_t)6 * BATCH * HDIM  * 4);
    float*  gates  = (float*)grab((size_t)NSPLIT * BATCH * NGATE * 4);
    __bf16* Wc1    = (__bf16*)grab((size_t)NGATE * KC1  * 2);
    __bf16* Wc2    = (__bf16*)grab((size_t)NGATE * KC23 * 2);
    __bf16* Wc3    = (__bf16*)grab((size_t)NGATE * KC23 * 2);
    __bf16* act1   = (__bf16*)grab((size_t)BATCH * KC1  * 2);
    __bf16* act2   = (__bf16*)grab((size_t)BATCH * KC23 * 2);
    __bf16* act3   = (__bf16*)grab((size_t)BATCH * KC23 * 2);

    float* h1 = states + 0 * BATCH * HDIM;
    float* c1 = states + 1 * BATCH * HDIM;
    float* h2 = states + 2 * BATCH * HDIM;
    float* c2 = states + 3 * BATCH * HDIM;
    float* h3 = states + 4 * BATCH * HDIM;
    float* c3 = states + 5 * BATCH * HDIM;

    // One-time: conv encoder + weight packing + state init
    conv1_kernel<<<(BATCH * 8  * T1   + 255) / 256, 256, 0, stream>>>(x,  w_c1, b_c1, e1);
    conv2_kernel<<<(BATCH * 16 * T2   + 255) / 256, 256, 0, stream>>>(e1, w_c2, b_c2, e2);
    conv3_kernel<<<(BATCH * LSEQ * 32 + 255) / 256, 256, 0, stream>>>(e2, w_c3, b_c3, enc);
    pack_wcat_kernel<<<2048, 256, 0, stream>>>(wih1, 96,   whh1, Wc1, KC1);
    pack_wcat_kernel<<<2048, 256, 0, stream>>>(wih2, 1056, whh2, Wc2, KC23);
    pack_wcat_kernel<<<2048, 256, 0, stream>>>(wih3, 1056, whh3, Wc3, KC23);
    init_state_kernel<<<512, 256, 0, stream>>>(x, states, act1, act2, act3);

    dim3 ggrid(NGATE / 32, NSPLIT);   // 128 N-blocks x 4 K-splits, 1 wave each
    for (int s = 0; s < NSTEP; ++s) {
        attend_kernel<<<dim3(3, BATCH), 256, 0, stream>>>(
            enc, h1, h2, h3, w_a1, w_a2, w_a3, act1, act2, act3);

        gemm_wmma_kernel<<<ggrid, 32, 0, stream>>>(act1, Wc1, gates, KC1);
        lstm_update_kernel<<<BATCH * HDIM / 256, 256, 0, stream>>>(
            gates, bih1, bhh1, h1, c1, act2, KC23, 32, act1, KC1, 96);

        gemm_wmma_kernel<<<ggrid, 32, 0, stream>>>(act2, Wc2, gates, KC23);
        lstm_update_kernel<<<BATCH * HDIM / 256, 256, 0, stream>>>(
            gates, bih2, bhh2, h2, c2, act3, KC23, 32, act2, KC23, 1056);

        gemm_wmma_kernel<<<ggrid, 32, 0, stream>>>(act3, Wc3, gates, KC23);
        lstm_update_kernel<<<BATCH * HDIM / 256, 256, 0, stream>>>(
            gates, bih3, bhh3, h3, c3, act3, KC23, 1056, (__bf16*)nullptr, 0, 0);

        out_kernel<<<BATCH, DOUT, 0, stream>>>(h3, wout, bout, out, act1, s);
    }
}
